// RNN_custom_25056839205499
// MI455X (gfx1250) — matrix operand order repaired and checked by reading, not compiled, not run on hardware
//
#include <hip/hip_runtime.h>
#include <cstddef>
#include <cstdint>

typedef _Float16 f16;
typedef __attribute__((ext_vector_type(16))) _Float16 v16h;
typedef __attribute__((ext_vector_type(8)))  _Float16 v8h;
typedef __attribute__((ext_vector_type(8)))  float    v8f;

#define HDIM   256
#define DIN    700
#define DINP   704
#define DOUT   20
#define DOUTP  32
#define NBATCH 128
#define TSTEPS 250

union V16 { v16h v; v8h h[2]; };
union V8F { v8f v; float f[8]; };

// A-fragment (16x32 f16): lane L holds row M = L%16; with g = L/16 the 16 halfs are
// K = kb + g*8 + {0..7} and K = kb + 16 + g*8 + {0..7}  (two 16B chunks).
static __device__ __forceinline__ v16h load_afrag(const f16* base, int ld, int row, int kb, int g) {
  V16 u;
  const f16* p = base + (size_t)row * ld + kb + g * 8;
  u.h[0] = *(const v8h*)(p);
  u.h[1] = *(const v8h*)(p + 16);
  return u.v;
}
// B-fragment (32x16 f16) from row-major W[n][k]: lane L holds col N = L%16,
// K = kb + g*8 + {0..7} and K = kb + 16 + g*8 + {0..7}  (two 16B chunks).
static __device__ __forceinline__ v16h load_bfrag(const f16* base, int ld, int row, int kb, int g) {
  V16 u;
  const f16* p = base + (size_t)row * ld + kb + g * 8;
  u.h[0] = *(const v8h*)(p);
  u.h[1] = *(const v8h*)(p + 16);
  return u.v;
}
static __device__ __forceinline__ v8f wmma_f16(v16h a, v16h b, v8f c) {
  return __builtin_amdgcn_wmma_f32_16x16x32_f16(false, a, false, b, (short)0, c, false, false);
}

// ---------------- weight convert (fp32 -> f16, with zero padding) ----------------
__global__ void cvt_pad_f16(const float* __restrict__ src, f16* __restrict__ dst,
                            int src_rows, int src_cols, int dst_cols, int total) {
  int i = blockIdx.x * blockDim.x + threadIdx.x;
  if (i >= total) return;
  int r = i / dst_cols, c = i - r * dst_cols;
  float v = (r < src_rows && c < src_cols) ? src[(size_t)r * src_cols + c] : 0.f;
  dst[i] = (f16)v;
}

// ---------------- phase 1: h1ff[t] = x_t @ Wi1^T + (bi1 + b11), fragment-major ----------------
// grid: 250*8 blocks (t, mt), 256 threads (8 waves, 2 n-tiles each)
__global__ __launch_bounds__(256) void ff_gemm(const float* __restrict__ x,
                                               const f16* __restrict__ Wi1h,
                                               const float* __restrict__ bi1,
                                               const float* __restrict__ b11,
                                               float* __restrict__ h1ff) {
  __shared__ f16 xs[16 * DINP];
  const int tid = threadIdx.x;
  const int t  = blockIdx.x >> 3;
  const int mt = blockIdx.x & 7;
  for (int i = tid; i < 16 * DINP; i += 256) {
    int r = i / DINP, k = i - r * DINP;
    float v = (k < DIN) ? x[((size_t)(mt * 16 + r) * TSTEPS + t) * DIN + k] : 0.f;
    xs[i] = (f16)v;
  }
  __syncthreads();
  const int w = tid >> 5, lane = tid & 31;
  const int lm = lane & 15, g = lane >> 4;
  V8F c[2];
#pragma unroll
  for (int j = 0; j < 2; ++j)
#pragma unroll
    for (int r = 0; r < 8; ++r) c[j].f[r] = 0.f;

  for (int kk = 0; kk < DINP / 32; ++kk) {
    int kb = kk * 32;
    v16h a = load_afrag(xs, DINP, lm, kb, g);
#pragma unroll
    for (int j = 0; j < 2; ++j) {
      v16h b = load_bfrag(Wi1h, DINP, (w * 2 + j) * 16 + lm, kb, g);
      c[j].v = wmma_f16(a, b, c[j].v);
    }
  }
#pragma unroll
  for (int j = 0; j < 2; ++j) {
    int nt = w * 2 + j;
    int n  = nt * 16 + lm;
    float bias = bi1[n] + b11[n];
    float* dst = h1ff + (((size_t)t * 128 + mt * 16 + nt) * 32 + lane) * 8;
#pragma unroll
    for (int r = 0; r < 8; ++r) dst[r] = c[j].f[r] + bias;
  }
}

// ---------------- phase 2: persistent recurrent kernel, one block per 16-row batch slice ------
__global__ __launch_bounds__(256) void snn_recurrent(
    const float* __restrict__ h1ff,
    const f16* __restrict__ W11h, const f16* __restrict__ W12h,
    const f16* __restrict__ W22h, const f16* __restrict__ Woh,
    const float* __restrict__ b12, const float* __restrict__ b22,
    const float* __restrict__ bo,
    const float* __restrict__ tadp1, const float* __restrict__ tadp2,
    const float* __restrict__ tm1, const float* __restrict__ tm2,
    const float* __restrict__ tmo,
    const float* __restrict__ mem1_0, const float* __restrict__ mem2_0,
    const float* __restrict__ memo_0, float* __restrict__ out) {
  __shared__ f16 sp1[16 * HDIM];     // spikes layer1 (binary, exact in f16)
  __shared__ f16 sp2[16 * HDIM];     // spikes layer2
  __shared__ float po[16 * DOUTP];   // sp2 @ Wo^T

  const int tid = threadIdx.x;
  const int bc  = blockIdx.x;        // batch slice: rows bc*16 .. bc*16+15
  const int w = tid >> 5, lane = tid & 31;
  const int lm = lane & 15, g = lane >> 4;

  for (int i = tid; i < 16 * HDIM; i += 256) { sp1[i] = (f16)0.f; sp2[i] = (f16)0.f; }

  // thread-owned state: tiles j=0,1 -> col n=(w*2+j)*16+lm, rows m=r+8*g (local)
  float mem1v[2][8], b1v[2][8], mem2v[2][8], b2v[2][8];
  float al1[2], ro1[2], al2[2], ro2[2], bias2[2];
  unsigned sm1[2] = {0u, 0u}, sm2[2] = {0u, 0u};
#pragma unroll
  for (int j = 0; j < 2; ++j) {
    int n = (w * 2 + j) * 16 + lm;
    al1[j] = expf(-1.f / tm1[n]);
    ro1[j] = expf(-1.f / tadp1[n]);
    al2[j] = expf(-1.f / tm2[n]);
    ro2[j] = expf(-1.f / tadp2[n]);
    bias2[j] = b12[n] + b22[n];
#pragma unroll
    for (int r = 0; r < 8; ++r) {
      int m = bc * 16 + r + 8 * g;
      mem1v[j][r] = mem1_0[(size_t)m * HDIM + n];
      mem2v[j][r] = mem2_0[(size_t)m * HDIM + n];
      b1v[j][r] = 0.01f;
      b2v[j][r] = 0.01f;
    }
  }
  // output-layer state owned by threads 0..15 (row = bc*16 + tid)
  float memo[DOUT], acc[DOUT], alo[DOUT], bov[DOUT];
  if (tid < 16) {
    int b = bc * 16 + tid;
    for (int o = 0; o < DOUT; ++o) {
      memo[o] = memo_0[b * DOUT + o];
      acc[o]  = 0.f;
      alo[o]  = expf(-1.f / tmo[o]);
      bov[o]  = bo[o];
    }
  }
  __syncthreads();

  for (int t = 0; t < TSTEPS; ++t) {
    // ---- phase A: C1 = h1ff[t] + sp1_old @ W11^T ----
    V8F c1[2];
#pragma unroll
    for (int j = 0; j < 2; ++j) {
      const float* src = h1ff + (((size_t)t * 128 + bc * 16 + (w * 2 + j)) * 32 + lane) * 8;
#pragma unroll
      for (int r = 0; r < 8; ++r) c1[j].f[r] = src[r];
    }
    for (int kk = 0; kk < HDIM / 32; ++kk) {
      v16h a = load_afrag(sp1, HDIM, lm, kk * 32, g);
#pragma unroll
      for (int j = 0; j < 2; ++j) {
        v16h bfr = load_bfrag(W11h, HDIM, (w * 2 + j) * 16 + lm, kk * 32, g);
        c1[j].v = wmma_f16(a, bfr, c1[j].v);
      }
    }
    __syncthreads();
    // ---- update layer 1, write sp1_new ----
#pragma unroll
    for (int j = 0; j < 2; ++j) {
      int n = (w * 2 + j) * 16 + lm;
      unsigned mnew = 0u;
#pragma unroll
      for (int r = 0; r < 8; ++r) {
        float sp = ((sm1[j] >> r) & 1u) ? 1.f : 0.f;
        b1v[j][r] = ro1[j] * b1v[j][r] + (1.f - ro1[j]) * sp;
        float Bv = 0.01f + 1.8f * b1v[j][r];
        mem1v[j][r] = mem1v[j][r] * al1[j] + (1.f - al1[j]) * c1[j].f[r] - Bv * sp;
        float s = (mem1v[j][r] - Bv) > 0.f ? 1.f : 0.f;
        mnew |= (s != 0.f ? 1u : 0u) << r;
        sp1[(r + 8 * g) * HDIM + n] = (f16)s;
      }
      sm1[j] = mnew;
    }
    __syncthreads();
    // ---- phase B: C2 = bias + sp1_new @ W12^T + sp2_old @ W22^T ; po = sp2_old @ Wo^T ----
    V8F c2[2];
#pragma unroll
    for (int j = 0; j < 2; ++j)
#pragma unroll
      for (int r = 0; r < 8; ++r) c2[j].f[r] = bias2[j];
    for (int kk = 0; kk < HDIM / 32; ++kk) {
      v16h a = load_afrag(sp1, HDIM, lm, kk * 32, g);
#pragma unroll
      for (int j = 0; j < 2; ++j) {
        v16h bfr = load_bfrag(W12h, HDIM, (w * 2 + j) * 16 + lm, kk * 32, g);
        c2[j].v = wmma_f16(a, bfr, c2[j].v);
      }
    }
    for (int kk = 0; kk < HDIM / 32; ++kk) {
      v16h a = load_afrag(sp2, HDIM, lm, kk * 32, g);
#pragma unroll
      for (int j = 0; j < 2; ++j) {
        v16h bfr = load_bfrag(W22h, HDIM, (w * 2 + j) * 16 + lm, kk * 32, g);
        c2[j].v = wmma_f16(a, bfr, c2[j].v);
      }
    }
    if (w < 2) {  // wave-uniform branch: EXEC stays all-ones for WMMA
      V8F cw;
#pragma unroll
      for (int r = 0; r < 8; ++r) cw.f[r] = 0.f;
      for (int kk = 0; kk < HDIM / 32; ++kk) {
        v16h a = load_afrag(sp2, HDIM, lm, kk * 32, g);
        v16h bfr = load_bfrag(Woh, HDIM, w * 16 + lm, kk * 32, g);
        cw.v = wmma_f16(a, bfr, cw.v);
      }
#pragma unroll
      for (int r = 0; r < 8; ++r) po[(r + 8 * g) * DOUTP + w * 16 + lm] = cw.f[r];
    }
    __syncthreads();
    // ---- update layer 2, write sp2_new; output layer + softmax accumulation ----
#pragma unroll
    for (int j = 0; j < 2; ++j) {
      int n = (w * 2 + j) * 16 + lm;
      unsigned mnew = 0u;
#pragma unroll
      for (int r = 0; r < 8; ++r) {
        float sp = ((sm2[j] >> r) & 1u) ? 1.f : 0.f;
        b2v[j][r] = ro2[j] * b2v[j][r] + (1.f - ro2[j]) * sp;
        float Bv = 0.01f + 1.8f * b2v[j][r];
        mem2v[j][r] = mem2v[j][r] * al2[j] + (1.f - al2[j]) * c2[j].f[r] - Bv * sp;
        float s = (mem2v[j][r] - Bv) > 0.f ? 1.f : 0.f;
        mnew |= (s != 0.f ? 1u : 0u) << r;
        sp2[(r + 8 * g) * HDIM + n] = (f16)s;
      }
      sm2[j] = mnew;
    }
    if (tid < 16) {
      float mx = -1e30f;
#pragma unroll
      for (int o = 0; o < DOUT; ++o) {
        float h = po[tid * DOUTP + o] + bov[o];
        memo[o] = memo[o] * alo[o] + (1.f - alo[o]) * h;
        mx = fmaxf(mx, memo[o]);
      }
      if (t > 10) {
        float e[DOUT];
        float ssum = 0.f;
#pragma unroll
        for (int o = 0; o < DOUT; ++o) { e[o] = __expf(memo[o] - mx); ssum += e[o]; }
        float inv = 1.f / ssum;
#pragma unroll
        for (int o = 0; o < DOUT; ++o) acc[o] += e[o] * inv;
      }
    }
    __syncthreads();
  }
  if (tid < 16) {
    int b = bc * 16 + tid;
    for (int o = 0; o < DOUT; ++o) out[b * DOUT + o] = acc[o];
  }
}

extern "C" void kernel_launch(void* const* d_in, const int* in_sizes, int n_in,
                              void* d_out, int out_size, void* d_ws, size_t ws_size,
                              hipStream_t stream) {
  (void)in_sizes; (void)n_in; (void)out_size; (void)ws_size;
  const float* x      = (const float*)d_in[0];
  const float* mem1_0 = (const float*)d_in[1];
  const float* mem2_0 = (const float*)d_in[2];
  const float* memo_0 = (const float*)d_in[3];
  const float* Wi1    = (const float*)d_in[4];
  const float* bi1    = (const float*)d_in[5];
  const float* W11    = (const float*)d_in[6];
  const float* b11    = (const float*)d_in[7];
  const float* W12    = (const float*)d_in[8];
  const float* b12    = (const float*)d_in[9];
  const float* W22    = (const float*)d_in[10];
  const float* b22    = (const float*)d_in[11];
  const float* Wo     = (const float*)d_in[12];
  const float* bo     = (const float*)d_in[13];
  const float* tadp1  = (const float*)d_in[14];
  const float* tadp2  = (const float*)d_in[15];
  const float* tm1    = (const float*)d_in[16];
  const float* tm2    = (const float*)d_in[17];
  const float* tmo    = (const float*)d_in[18];

  char* ws = (char*)d_ws;
  size_t off = 0;
  float* h1ff = (float*)(ws + off); off += (size_t)TSTEPS * NBATCH * HDIM * sizeof(float);
  f16* Wi1h = (f16*)(ws + off);     off += (size_t)HDIM  * DINP * sizeof(f16);
  f16* W11h = (f16*)(ws + off);     off += (size_t)HDIM  * HDIM * sizeof(f16);
  f16* W12h = (f16*)(ws + off);     off += (size_t)HDIM  * HDIM * sizeof(f16);
  f16* W22h = (f16*)(ws + off);     off += (size_t)HDIM  * HDIM * sizeof(f16);
  f16* Woh  = (f16*)(ws + off);     off += (size_t)DOUTP * HDIM * sizeof(f16);

  cvt_pad_f16<<<(HDIM * DINP + 255) / 256, 256, 0, stream>>>(Wi1, Wi1h, HDIM, DIN, DINP, HDIM * DINP);
  cvt_pad_f16<<<(HDIM * HDIM + 255) / 256, 256, 0, stream>>>(W11, W11h, HDIM, HDIM, HDIM, HDIM * HDIM);
  cvt_pad_f16<<<(HDIM * HDIM + 255) / 256, 256, 0, stream>>>(W12, W12h, HDIM, HDIM, HDIM, HDIM * HDIM);
  cvt_pad_f16<<<(HDIM * HDIM + 255) / 256, 256, 0, stream>>>(W22, W22h, HDIM, HDIM, HDIM, HDIM * HDIM);
  cvt_pad_f16<<<(DOUTP * HDIM + 255) / 256, 256, 0, stream>>>(Wo, Woh, DOUT, HDIM, HDIM, DOUTP * HDIM);

  ff_gemm<<<TSTEPS * 8, 256, 0, stream>>>(x, Wi1h, bi1, b11, h1ff);

  snn_recurrent<<<8, 256, 0, stream>>>(h1ff, W11h, W12h, W22h, Woh, b12, b22, bo,
                                       tadp1, tadp2, tm1, tm2, tmo,
                                       mem1_0, mem2_0, memo_0, (float*)d_out);
}
